// LinearAttention_9569187136035
// MI455X (gfx1250) — matrix operand order repaired and checked by reading, not compiled, not run on hardware
//
#include <hip/hip_runtime.h>
#include <hip/hip_bf16.h>

// ---- fixed problem shape (from reference setup_inputs) ----
constexpr int B = 4, S = 4096, E = 1024, H = 16;
constexpr int BS = B * S;           // 16384 rows
constexpr int BSE = BS * E;         // 16,777,216 elements

typedef __attribute__((ext_vector_type(16))) __bf16 v16bf;
typedef __attribute__((ext_vector_type(8)))  __bf16 v8bf;
typedef __attribute__((ext_vector_type(8)))  float  v8f;
typedef __attribute__((ext_vector_type(4)))  unsigned int u32x4;
typedef __attribute__((ext_vector_type(8)))  int          i32x8;
typedef __attribute__((ext_vector_type(4)))  int          i32x4;

// =====================================================================
// TDM: DMA a 2D bf16 tile (tileRows x tileCols, row stride strideElems)
// from global memory into LDS at byte offset ldsOff. D# packed per
// CDNA5 ISA ch.8 (group0/group1 bitfields); groups 2/3 unused (2D).
// This toolchain's builtin takes 6 args (g0, g1, g2, g3, g4, cpol).
// =====================================================================
__device__ inline void tdm_load_tile_bf16(const __bf16* gsrc, unsigned ldsOff,
                                          int tileRows, int tileCols, int strideElems) {
  unsigned long long ga = (unsigned long long)(uintptr_t)gsrc;
  u32x4 g0;
  g0[0] = 1u;                                                 // count=1, user descriptor
  g0[1] = ldsOff;                                             // lds_addr (bytes)
  g0[2] = (unsigned)ga;                                       // global_addr[31:0]
  g0[3] = (unsigned)((ga >> 32) & 0x01ffffffu) | (2u << 30);  // addr[56:32] | type=2
  unsigned td0 = (unsigned)tileCols;                          // tensor_dim0
  unsigned td1 = (unsigned)tileRows;                          // tensor_dim1
  unsigned long long st0 = (unsigned long long)strideElems;   // tensor_dim0_stride
  i32x8 g1;
  g1[0] = (int)(1u << 16);                                    // wg_mask=0, data_size=2B
  g1[1] = (int)((td0 & 0xffffu) << 16);                       // tensor_dim0[15:0] @ 63:48
  g1[2] = (int)((td0 >> 16) | ((td1 & 0xffffu) << 16));       // td0 hi | td1 lo
  g1[3] = (int)((td1 >> 16) | ((unsigned)tileCols << 16));    // td1 hi | tile_dim0
  g1[4] = (int)((unsigned)tileRows & 0xffffu);                // tile_dim1 | tile_dim2=0
  g1[5] = (int)(unsigned)(st0 & 0xffffffffu);                 // stride0[31:0]
  g1[6] = (int)(unsigned)((st0 >> 32) & 0xffffu);             // stride0[47:32]
  g1[7] = 0;
  i32x4 z4 = {0, 0, 0, 0};
  i32x8 z8 = {0, 0, 0, 0, 0, 0, 0, 0};
  __builtin_amdgcn_tensor_load_to_lds(g0, g1, z4, z4, z8, 0);
}

// =====================================================================
// elementwise: mask + cast q,k,v to bf16
// =====================================================================
__global__ void cast_mask_inputs(const float* __restrict__ q, const float* __restrict__ k,
                                 const float* __restrict__ v,
                                 const int* __restrict__ qmask, const int* __restrict__ kmask,
                                 __bf16* __restrict__ qb, __bf16* __restrict__ kb,
                                 __bf16* __restrict__ vb, int n, int e) {
  int i = blockIdx.x * blockDim.x + threadIdx.x;
  if (i >= n) return;
  int row = i / e;                       // b*S + s
  bool qm = qmask[row] != 0;
  bool km = kmask[row] != 0;
  qb[i] = (__bf16)(qm ? q[i] : -1e9f);
  kb[i] = (__bf16)(km ? k[i] : -1e9f);
  vb[i] = (__bf16)(km ? v[i] : 0.0f);
}

__global__ void cast_f32_bf16(const float* __restrict__ src, __bf16* __restrict__ dst, int n) {
  int i = blockIdx.x * blockDim.x + threadIdx.x;
  if (i < n) dst[i] = (__bf16)src[i];
}

// =====================================================================
// Tiled GEMM: C(M,N) f32 = A(M,K)bf16 @ W(N,K)bf16^T + bias, optional
// row mask. Block tile 128x128, BK=32, TDM double-buffered LDS staging,
// 8 waves, each computing a 64x32 sub-tile (4x2 WMMA accumulators).
// Requires M%128==0, N%128==0, K%32==0.
// =====================================================================
__global__ void __launch_bounds__(256)
gemm_bf16_wmma(const __bf16* __restrict__ A, const __bf16* __restrict__ W,
               const float* __restrict__ bias, float* __restrict__ C,
               int M, int N, int K, const int* __restrict__ rowmask) {
  constexpr int BM = 128, BN = 128, BK = 32;
  __shared__ __align__(32) __bf16 Abuf[2][BM * BK];   // [row][k] row-major
  __shared__ __align__(32) __bf16 Bbuf[2][BN * BK];   // [n][k]  row-major (= W rows)

  const int lane   = threadIdx.x & 31;
  const int waveId = threadIdx.x >> 5;
  const int l  = lane & 15;
  const int hi = lane >> 4;
  const int wm = waveId >> 2;          // 0..1  -> 64-row slab
  const int wn = waveId & 3;           // 0..3  -> 32-col slab

  const int tilesN = N / BN;
  const int rowBase = (blockIdx.x / tilesN) * BM;
  const int colBase = (blockIdx.x % tilesN) * BN;

  const __bf16* Ag = A + (size_t)rowBase * K;
  const __bf16* Wg = W + (size_t)colBase * K;

  // ---- prologue: DMA first K-slab into buffer 0 ----
  if (waveId == 0) {
    tdm_load_tile_bf16(Ag, (unsigned)(uintptr_t)&Abuf[0][0], BM, BK, K);
    tdm_load_tile_bf16(Wg, (unsigned)(uintptr_t)&Bbuf[0][0], BN, BK, K);
    __builtin_amdgcn_s_wait_tensorcnt(0);
  }
  __syncthreads();

  v8f acc[4][2] = {};
  const int kSteps = K / BK;
  for (int kt = 0; kt < kSteps; ++kt) {
    const int cur = kt & 1;
    const int nxt = cur ^ 1;
    if (kt + 1 < kSteps && waveId == 0) {
      const int k0 = (kt + 1) * BK;
      tdm_load_tile_bf16(Ag + k0, (unsigned)(uintptr_t)&Abuf[nxt][0], BM, BK, K);
      tdm_load_tile_bf16(Wg + k0, (unsigned)(uintptr_t)&Bbuf[nxt][0], BN, BK, K);
    }

    const __bf16* Ab = &Abuf[cur][0];
    const __bf16* Bb = &Bbuf[cur][0];

    // A-frags: lane holds row M=l; hi=0 -> K={0..7,16..23}; hi=1 -> +8
    v16bf fa[4];
#pragma unroll
    for (int f = 0; f < 4; ++f) {
      const __bf16* p = Ab + (wm * 64 + f * 16 + l) * BK + hi * 8;
      v8bf lo = *(const v8bf*)p;
      v8bf hi8 = *(const v8bf*)(p + 16);
      fa[f] = __builtin_shufflevector(lo, hi8, 0, 1, 2, 3, 4, 5, 6, 7,
                                      8, 9, 10, 11, 12, 13, 14, 15);
    }
    // B-frags: lane holds col N=l; hi=0 -> K={0..7,16..23}; hi=1 -> +8
    v16bf fb[2];
#pragma unroll
    for (int g = 0; g < 2; ++g) {
      const __bf16* q = Bb + (wn * 32 + g * 16 + l) * BK + hi * 8;
      v8bf blo = *(const v8bf*)q;
      v8bf bhi = *(const v8bf*)(q + 16);
      fb[g] = __builtin_shufflevector(blo, bhi, 0, 1, 2, 3, 4, 5, 6, 7,
                                      8, 9, 10, 11, 12, 13, 14, 15);
    }

#pragma unroll
    for (int f = 0; f < 4; ++f)
#pragma unroll
      for (int g = 0; g < 2; ++g)
        acc[f][g] = __builtin_amdgcn_wmma_f32_16x16x32_bf16(
            false, fa[f], false, fb[g], (short)0, acc[f][g], false, false);

    if (kt + 1 < kSteps) {
      if (waveId == 0) __builtin_amdgcn_s_wait_tensorcnt(0);
      __syncthreads();
    }
  }

  // ---- epilogue ----
#pragma unroll
  for (int f = 0; f < 4; ++f) {
#pragma unroll
    for (int g = 0; g < 2; ++g) {
      const int col = colBase + wn * 32 + g * 16 + l;
      const float bv = bias ? bias[col] : 0.0f;
#pragma unroll
      for (int r = 0; r < 8; ++r) {
        const int row = rowBase + wm * 64 + f * 16 + hi * 8 + r;
        float val = acc[f][g][r] + bv;
        if (rowmask) val *= (rowmask[row] != 0) ? 1.0f : 0.0f;
        C[(size_t)row * N + col] = val;
      }
    }
  }
}

// =====================================================================
// softmax over D=64 within each head row; one wave per head row.
// =====================================================================
__global__ void softmax64_bf16(const float* __restrict__ X, __bf16* __restrict__ Y, int rows) {
  int wave = blockIdx.x * (blockDim.x >> 5) + (threadIdx.x >> 5);
  if (wave >= rows) return;
  int lane = threadIdx.x & 31;
  const size_t base = (size_t)wave * 64;
  float x0 = X[base + lane];
  float x1 = X[base + 32 + lane];
  float m = fmaxf(x0, x1);
#pragma unroll
  for (int i = 16; i >= 1; i >>= 1) m = fmaxf(m, __shfl_xor(m, i, 32));
  float e0 = __expf(x0 - m), e1 = __expf(x1 - m);
  float s = e0 + e1;
#pragma unroll
  for (int i = 16; i >= 1; i >>= 1) s += __shfl_xor(s, i, 32);
  float inv = 1.0f / s;
  Y[base + lane]      = (__bf16)(e0 * inv);
  Y[base + 32 + lane] = (__bf16)(e1 * inv);
}

// =====================================================================
// k_cumsum[b,h,d] = sum_s kh[b,s,h*64+d]; one block per (b,h)
// =====================================================================
__global__ void kcumsum_kernel(const __bf16* __restrict__ ks, float* __restrict__ kcum) {
  int bh = blockIdx.x;
  int b = bh >> 4, h = bh & 15;
  __shared__ float sdata[256];
  int t = threadIdx.x;
  int d = t & 63, part = t >> 6;
  const __bf16* base = ks + (size_t)b * S * E + h * 64 + d;
  float sum = 0.0f;
  for (int s = part; s < S; s += 4) sum += (float)base[(size_t)s * E];
  sdata[t] = sum;
  __syncthreads();
  if (t < 64)
    kcum[bh * 64 + t] = sdata[t] + sdata[64 + t] + sdata[128 + t] + sdata[192 + t];
}

// =====================================================================
// context[b,h] (64x64, bf16) = kh^T @ vh, K = S = 4096. One block per
// (b,h); 16 waves, each owns one 16x16 tile. A = kh^T -> strided frags.
// =====================================================================
__global__ void context_wmma(const __bf16* __restrict__ ks, const __bf16* __restrict__ vs,
                             __bf16* __restrict__ ctxb) {
  int bh = blockIdx.x;
  int b = bh >> 4, h = bh & 15;
  int waveId = threadIdx.x >> 5, lane = threadIdx.x & 31;
  int d0 = (waveId >> 2) * 16;     // M tile (d index)
  int e0 = (waveId & 3) * 16;      // N tile (e index)
  int l = lane & 15, hi = lane >> 4;
  const __bf16* kbase = ks + (size_t)b * S * E + h * 64;
  const __bf16* vbase = vs + (size_t)b * S * E + h * 64;

  v8f acc = {};
  for (int s0 = 0; s0 < S; s0 += 32) {
    v16bf a, bb;
    const __bf16* pa = kbase + (size_t)s0 * E + d0 + l;
    __builtin_prefetch(pa + 32 * E, 0, 0);
#pragma unroll
    for (int i = 0; i < 8; ++i) {
      a[i]     = pa[(size_t)(hi * 8 + i) * E];
      a[8 + i] = pa[(size_t)(16 + hi * 8 + i) * E];
    }
    const __bf16* pb = vbase + (size_t)s0 * E + e0 + l;
#pragma unroll
    for (int i = 0; i < 8; ++i) {
      bb[i]     = pb[(size_t)(hi * 8 + i) * E];
      bb[8 + i] = pb[(size_t)(16 + hi * 8 + i) * E];
    }
    acc = __builtin_amdgcn_wmma_f32_16x16x32_bf16(false, a, false, bb, (short)0, acc,
                                                  false, false);
  }
  __bf16* cb = ctxb + (size_t)bh * 64 * 64;
#pragma unroll
  for (int r = 0; r < 8; ++r) {
    int d = d0 + hi * 8 + r, e = e0 + l;
    cb[d * 64 + e] = (__bf16)acc[r];
  }
}

// =====================================================================
// a_t[b,h,s] = 1 / sum_d qh[b,s,h*64+d] * kcum[b,h,d]
// =====================================================================
__global__ void at_kernel(const __bf16* __restrict__ qs, const float* __restrict__ kcum,
                          float* __restrict__ a_t) {
  int idx = blockIdx.x * blockDim.x + threadIdx.x;     // over B*H*S
  if (idx >= B * H * S) return;
  int s = idx % S, bh = idx / S;
  int b = bh / H, h = bh % H;
  const __bf16* qrow = qs + ((size_t)(b * S + s)) * E + h * 64;
  const float* kc = kcum + bh * 64;
  float dot = 0.0f;
#pragma unroll 8
  for (int d = 0; d < 64; ++d) dot += (float)qrow[d] * kc[d];
  a_t[idx] = 1.0f / dot;
}

// =====================================================================
// pre[b,s,h*64+e] = (qh @ context)[s,e] * a_t[b,h,s] + qh[s,e]   (bf16)
// =====================================================================
__global__ void outstage_wmma(const __bf16* __restrict__ qs, const __bf16* __restrict__ ctxb,
                              const float* __restrict__ a_t, __bf16* __restrict__ pre) {
  const int eTiles = 4, sTiles = S >> 4, tilesPerBH = eTiles * sTiles;
  int wave = blockIdx.x * (blockDim.x >> 5) + (threadIdx.x >> 5);
  if (wave >= B * H * tilesPerBH) return;
  int lane = threadIdx.x & 31;
  int bh = wave / tilesPerBH, rem = wave % tilesPerBH;
  int s0 = (rem / eTiles) * 16, e0 = (rem % eTiles) * 16;
  int b = bh >> 4, h = bh & 15;
  int l = lane & 15, hi = lane >> 4;
  const __bf16* qbase = qs + (size_t)b * S * E + h * 64;
  const __bf16* cbase = ctxb + (size_t)bh * 4096;

  v8f acc = {};
#pragma unroll
  for (int k0 = 0; k0 < 64; k0 += 32) {
    v16bf a, bb;
    const __bf16* pa = qbase + (size_t)(s0 + l) * E + k0 + hi * 8;
#pragma unroll
    for (int i = 0; i < 8; ++i) { a[i] = pa[i]; a[8 + i] = pa[16 + i]; }
    const __bf16* pb = cbase + (size_t)(k0 + hi * 8) * 64 + e0 + l;
#pragma unroll
    for (int i = 0; i < 8; ++i) { bb[i] = pb[(size_t)i * 64]; bb[8 + i] = pb[(size_t)(16 + i) * 64]; }
    acc = __builtin_amdgcn_wmma_f32_16x16x32_bf16(false, a, false, bb, (short)0, acc,
                                                  false, false);
  }
  const float* at = a_t + (size_t)bh * S;
  int e = e0 + l;
#pragma unroll
  for (int r = 0; r < 8; ++r) {
    int s = s0 + hi * 8 + r;
    float qv = (float)qbase[(size_t)s * E + e];
    pre[((size_t)b * S + s) * E + h * 64 + e] = (__bf16)(acc[r] * at[s] + qv);
  }
}

// =====================================================================
// host-side launch
// =====================================================================
extern "C" void kernel_launch(void* const* d_in, const int* in_sizes, int n_in,
                              void* d_out, int out_size, void* d_ws, size_t ws_size,
                              hipStream_t stream) {
  (void)in_sizes; (void)n_in; (void)out_size; (void)ws_size;
  const float* q  = (const float*)d_in[0];
  const float* k  = (const float*)d_in[1];
  const float* v  = (const float*)d_in[2];
  const int* qmask = (const int*)d_in[3];   // (B,Sq,1) bool -> int per harness rule
  const int* kmask = (const int*)d_in[4];
  const float* Wq = (const float*)d_in[5];  const float* bq = (const float*)d_in[6];
  const float* Wk = (const float*)d_in[7];  const float* bk = (const float*)d_in[8];
  const float* Wv = (const float*)d_in[9];  const float* bv = (const float*)d_in[10];
  const float* Wp = (const float*)d_in[11]; const float* bp = (const float*)d_in[12];
  float* out = (float*)d_out;

  // ---- workspace carve-out ----
  size_t off = 0;
  auto take = [&](size_t bytes) -> void* {
    void* p = (char*)d_ws + off;
    off += (bytes + 255) & ~(size_t)255;
    return p;
  };
  __bf16* qb   = (__bf16*)take((size_t)BSE * 2);
  __bf16* kb   = (__bf16*)take((size_t)BSE * 2);
  __bf16* vb   = (__bf16*)take((size_t)BSE * 2);
  __bf16* wqb  = (__bf16*)take((size_t)E * E * 2);
  __bf16* wkb  = (__bf16*)take((size_t)E * E * 2);
  __bf16* wvb  = (__bf16*)take((size_t)E * E * 2);
  __bf16* wpb  = (__bf16*)take((size_t)E * E * 2);
  float*  proj = (float*)take((size_t)BSE * 4);
  float*  kcum = (float*)take((size_t)B * H * 64 * 4);
  __bf16* ctxb = (__bf16*)take((size_t)B * H * 4096 * 2);
  float*  a_t  = (float*)take((size_t)B * H * S * 4);
  __bf16* pre  = (__bf16*)take((size_t)BSE * 2);

  const int nW = E * E;
  const int gemmBlocks = (BS / 128) * (E / 128);   // 128 * 8 = 1024

  // 1) mask + bf16 casts
  cast_mask_inputs<<<(BSE + 255) / 256, 256, 0, stream>>>(q, k, v, qmask, kmask,
                                                          qb, kb, vb, BSE, E);
  cast_f32_bf16<<<(nW + 255) / 256, 256, 0, stream>>>(Wq, wqb, nW);
  cast_f32_bf16<<<(nW + 255) / 256, 256, 0, stream>>>(Wk, wkb, nW);
  cast_f32_bf16<<<(nW + 255) / 256, 256, 0, stream>>>(Wv, wvb, nW);
  cast_f32_bf16<<<(nW + 255) / 256, 256, 0, stream>>>(Wp, wpb, nW);

  // 2) q projection -> softmax (qs overwrites qb)
  gemm_bf16_wmma<<<gemmBlocks, 256, 0, stream>>>(qb, wqb, bq, proj, BS, E, E, nullptr);
  softmax64_bf16<<<(BS * H) / 8, 256, 0, stream>>>(proj, qb, BS * H);

  // 3) k projection -> softmax (ks overwrites kb)
  gemm_bf16_wmma<<<gemmBlocks, 256, 0, stream>>>(kb, wkb, bk, proj, BS, E, E, nullptr);
  softmax64_bf16<<<(BS * H) / 8, 256, 0, stream>>>(proj, kb, BS * H);

  // 4) v projection (no softmax) -> bf16 (vs overwrites vb)
  gemm_bf16_wmma<<<gemmBlocks, 256, 0, stream>>>(vb, wvb, bv, proj, BS, E, E, nullptr);
  cast_f32_bf16<<<(BSE + 255) / 256, 256, 0, stream>>>(proj, vb, BSE);

  // 5) k column sum, context, a_t
  kcumsum_kernel<<<B * H, 256, 0, stream>>>(kb, kcum);
  context_wmma<<<B * H, 512, 0, stream>>>(kb, vb, ctxb);
  at_kernel<<<(B * H * S + 255) / 256, 256, 0, stream>>>(qb, kcum, a_t);

  // 6) out = qh@ctx * a_t + qh  -> pre (bf16, merged heads layout)
  const int outWaves = B * H * (S / 16) * 4;     // 65536
  outstage_wmma<<<outWaves / 8, 256, 0, stream>>>(qb, ctxb, a_t, pre);

  // 7) final projection + bias, multiply by q mask
  gemm_bf16_wmma<<<gemmBlocks, 256, 0, stream>>>(pre, wpb, bp, out, BS, E, E, qmask);
}
